// Shift_gcn_73753178407267
// MI455X (gfx1250) — matrix operand order repaired and checked by reading, not compiled, not run on hardware
//
#include <hip/hip_runtime.h>

typedef __bf16 bf16;
typedef float  f32x8  __attribute__((ext_vector_type(8)));
typedef bf16   bf16x16 __attribute__((ext_vector_type(16)));

namespace {

constexpr int NB = 32, TT = 128, V = 25;
constexpr int NT = NB * TT;          // 4096 frames
constexpr int R  = NT * V;           // 102400 rows (n*t*V)

// ---------- workspace layout (bytes, 256-aligned) ----------
constexpr size_t OFF_X   = 0;                         // f32  (R,128)   layer input
constexpr size_t OFF_XB  = OFF_X   + (size_t)R*128*4; // bf16 (R,128)   layer input
constexpr size_t OFF_XS  = OFF_XB  + (size_t)R*128*2; // bf16 (R,128)   shifted/masked A; aliases XG
constexpr size_t OFF_Y   = OFF_XS  + (size_t)R*128*2; // f32  (R,128)   linear out
constexpr size_t OFF_D   = OFF_Y   + (size_t)R*128*4; // f32  (R,128)   down conv; aliases OUT
constexpr size_t OFF_XR  = OFF_D   + (size_t)R*128*4; // bf16 (R,128)   relu'd residual; aliases G1b
constexpr size_t OFF_G   = OFF_XR  + (size_t)R*128*2; // f32  (NT,25,25) attention
constexpr size_t OFF_ST1 = OFF_G   + (size_t)NT*V*V*4;
constexpr size_t OFF_ST2 = OFF_ST1 + 2*3200*4;
constexpr size_t OFF_ST3 = OFF_ST2 + 2*128*4 + 256;

struct Pack32 { uint4 lo, hi; };

__device__ __forceinline__ bf16x16 frag_ld(const bf16* lo, const bf16* hi) {
  Pack32 p;
  p.lo = *reinterpret_cast<const uint4*>(lo);
  p.hi = *reinterpret_cast<const uint4*>(hi);
  return __builtin_bit_cast(bf16x16, p);
}

// ---------------- pack x0 (n,c,t,V) -> ntvc f32 + bf16 ----------------
__global__ void pack_x0_k(const float* __restrict__ x0,
                          float* __restrict__ X, bf16* __restrict__ Xb) {
  int idx = blockIdx.x * 256 + threadIdx.x;
  if (idx >= R * 64) return;
  int c = idx & 63, r = idx >> 6;
  int v = r % V, f = r / V;
  int t = f & (TT - 1), n = f >> 7;
  float val = x0[(((size_t)(n * 64 + c)) * TT + t) * V + v];
  X[idx] = val;  Xb[idx] = (bf16)val;
}

// ---------------- channel shift + tanh feature mask -> bf16 ----------------
__global__ void shift_mask_k(const float* __restrict__ X, const float* __restrict__ fm,
                             bf16* __restrict__ XS, int cin) {
  int idx = blockIdx.x * 256 + threadIdx.x;
  if (idx >= R * cin) return;
  int c = idx % cin, r = idx / cin;
  int v = r % V, f = r / V;
  int vs = (v + c) % V;                                   // input shift: src row (v+c)%V, same channel
  float val = X[((size_t)(f * V + vs)) * cin + c] * (tanhf(fm[v * cin + c]) + 1.0f);
  XS[idx] = (bf16)val;
}

// ---------------- generic bf16 WMMA GEMM: C = A*Bt' [+ A2*Bt2'] + bias ----------------
// Bt LDS layout: Bt[nn][k] (contiguous k). wRowMajorOC=1 -> Bt[nn][k]=W[nn*K+k] (conv o,c weights)
//                                     else -> Bt[nn][k]=W[k*N+nn]              (lw (cin,cout))
__global__ __launch_bounds__(256)
void gemm_bf16_k(const bf16* __restrict__ A, const bf16* __restrict__ A2,
                 const float* __restrict__ W, const float* __restrict__ W2,
                 int wRowMajorOC, int w2RowMajorOC,
                 const float* __restrict__ bias,
                 float* __restrict__ outF, bf16* __restrict__ outB,
                 int M, int N, int K) {
  __shared__ bf16 Bt[2][16 * 128];
  const int tid = threadIdx.x;
  const int nt0 = blockIdx.y * 16;
  for (int i = tid; i < 16 * K; i += 256) {
    int nn = i / K, k = i - nn * K;
    int gn = nt0 + nn;
    Bt[0][i] = (bf16)(wRowMajorOC ? W[(size_t)gn * K + k] : W[(size_t)k * N + gn]);
    if (W2)
      Bt[1][i] = (bf16)(w2RowMajorOC ? W2[(size_t)gn * K + k] : W2[(size_t)k * N + gn]);
  }
  __syncthreads();
  const int wave = tid >> 5, lane = tid & 31;
  const int half = lane >> 4, l16 = lane & 15;
  const int kbA = half * 8, kbB = half * 16;
  const int mtile = blockIdx.x * 8 + wave;
  const size_t mrow = (size_t)(mtile * 16 + l16);

  f32x8 acc = {0.f, 0.f, 0.f, 0.f, 0.f, 0.f, 0.f, 0.f};
  {
    const bf16* arow = A + mrow * K;
    const bf16* brow = &Bt[0][l16 * K];
    for (int k0 = 0; k0 < K; k0 += 32) {
      bf16x16 af = frag_ld(arow + k0 + kbA, arow + k0 + kbA + 16);
      bf16x16 bfv = frag_ld(brow + k0 + kbA, brow + k0 + kbA + 16);
      acc = __builtin_amdgcn_wmma_f32_16x16x32_bf16(false, af, false, bfv,
                                                    (short)0, acc, false, false);
    }
  }
  if (A2) {
    const bf16* arow = A2 + mrow * K;
    const bf16* brow = &Bt[1][l16 * K];
    for (int k0 = 0; k0 < K; k0 += 32) {
      bf16x16 af = frag_ld(arow + k0 + kbA, arow + k0 + kbA + 16);
      bf16x16 bfv = frag_ld(brow + k0 + kbA, brow + k0 + kbA + 16);
      acc = __builtin_amdgcn_wmma_f32_16x16x32_bf16(false, af, false, bfv,
                                                    (short)0, acc, false, false);
    }
  }
  const int gn = nt0 + l16;
  const float bv = bias ? bias[gn] : 0.f;
  const int mbase = mtile * 16 + half * 8;
  for (int r2 = 0; r2 < 8; ++r2) {
    float val = acc[r2] + bv;
    size_t o = (size_t)(mbase + r2) * N + gn;
    if (outF) outF[o] = val;
    if (outB) outB[o] = (bf16)val;
  }
}

// ---------------- per-frame attention logits: G1(25x32) x G2(25x32)^T via WMMA ----------------
__global__ __launch_bounds__(128)
void glogits_k(const bf16* __restrict__ G1, const bf16* __restrict__ G2,
               float* __restrict__ GL) {
  const int tid = threadIdx.x;
  const int wave = tid >> 5, lane = tid & 31;
  const int mt = wave & 1, nt2 = wave >> 1;
  const int half = lane >> 4, l16 = lane & 15;
  const int f = blockIdx.x;
  const size_t base = (size_t)f * V;
  const bf16* arow = G1 + (base + mt * 16 + l16) * 32 + half * 8;
  const bf16* brow = G2 + (base + nt2 * 16 + l16) * 32 + half * 8;
  bf16x16 af = frag_ld(arow, arow + 16);
  bf16x16 bfv = frag_ld(brow, brow + 16);
  f32x8 acc = {0.f, 0.f, 0.f, 0.f, 0.f, 0.f, 0.f, 0.f};
  acc = __builtin_amdgcn_wmma_f32_16x16x32_bf16(false, af, false, bfv,
                                                (short)0, acc, false, false);
  const int nn = nt2 * 16 + l16;
  if (nn < V) {
    for (int r2 = 0; r2 < 8; ++r2) {
      int u = mt * 16 + half * 8 + r2;
      if (u < V) GL[(size_t)f * (V * V) + u * V + nn] = acc[r2];
    }
  }
}

// ---------------- softmax over last-25 ----------------
__global__ void softmax25_k(float* __restrict__ g, int nRows) {
  int i = blockIdx.x * 256 + threadIdx.x;
  if (i >= nRows) return;
  float* p = g + (size_t)i * V;
  float v[V], m = -3.0e38f;
  for (int j = 0; j < V; ++j) { v[j] = p[j]; m = fmaxf(m, v[j]); }
  float s = 0.f;
  for (int j = 0; j < V; ++j) { v[j] = expf(v[j] - m); s += v[j]; }
  float inv = 1.0f / s;
  for (int j = 0; j < V; ++j) p[j] = v[j] * inv;
}

// ---------------- column stats (mean/var over samples) via atomics ----------------
__global__ void colstats_k(const float* __restrict__ src, float* __restrict__ stats,
                           int F, int nS) {
  int f = blockIdx.x * 256 + threadIdx.x;
  if (f >= F) return;
  int r0 = blockIdx.y * 512;
  int r1 = r0 + 512 < nS ? r0 + 512 : nS;
  float s = 0.f, s2 = 0.f;
  for (int r = r0; r < r1; ++r) {
    float val = src[(size_t)r * F + f];
    s += val; s2 += val * val;
  }
  atomicAdd(&stats[f], s);
  atomicAdd(&stats[F + f], s2);
}

__global__ void stats_fin_k(float* __restrict__ stats, int F, float invN) {
  int f = blockIdx.x * 256 + threadIdx.x;
  if (f >= F) return;
  float m = stats[f] * invN;
  float var = stats[F + f] * invN - m * m;
  stats[f] = m;
  stats[F + f] = rsqrtf(var + 1e-5f);
}

// ---------------- output shift + BN1d + residual(identity or BN2d(conv)) + relu -> bf16 ----------------
__global__ void bn_shift_res_k(const float* __restrict__ Y, const float* __restrict__ st,
                               const float* __restrict__ bn_g, const float* __restrict__ bn_b,
                               const float* __restrict__ Xid,   // identity residual (or null)
                               const float* __restrict__ D,     // down conv out (or null)
                               const float* __restrict__ dst,   // down chan stats (or null)
                               const float* __restrict__ dbn_g, const float* __restrict__ dbn_b,
                               bf16* __restrict__ XR, int cout) {
  int idx = blockIdx.x * 256 + threadIdx.x;
  if (idx >= R * cout) return;
  int c = idx % cout, r = idx / cout;
  int v = r % V, f = r / V;
  int vs = v - (c % V); if (vs < 0) vs += V;              // output shift: src row (v-c)%V
  int sfeat = vs * cout + c;                              // column of Y feeding this feature
  int F = V * cout;
  float y = Y[((size_t)(f * V + vs)) * cout + c];
  int fo = v * cout + c;
  float z = (y - st[sfeat]) * st[F + sfeat] * bn_g[fo] + bn_b[fo];
  float dn;
  if (D) dn = (D[idx] - dst[c]) * dst[cout + c] * dbn_g[c] + dbn_b[c];
  else   dn = Xid[idx];
  float x = z + dn;
  XR[idx] = (bf16)fmaxf(x, 0.f);
}

// ---------------- per-frame xg = G(25x25) x XR(25xcout), VALU in LDS ----------------
__global__ __launch_bounds__(256)
void xg_k(const float* __restrict__ G, const bf16* __restrict__ XR,
          bf16* __restrict__ XG, int cout) {
  __shared__ float Gs[V * V];
  __shared__ float Xs[V * 128];
  const int f = blockIdx.x, tid = threadIdx.x;
  for (int i = tid; i < V * V; i += 256) Gs[i] = G[(size_t)f * (V * V) + i];
  for (int i = tid; i < V * cout; i += 256) Xs[i] = (float)XR[(size_t)f * V * cout + i];
  __syncthreads();
  for (int o = tid; o < V * cout; o += 256) {
    int u = o / cout, c = o - u * cout;
    float s = 0.f;
    for (int v2 = 0; v2 < V; ++v2) s += Gs[u * V + v2] * Xs[v2 * cout + c];
    XG[(size_t)f * V * cout + o] = (bf16)s;
  }
}

// ---------------- final BN2d + relu -> next-layer f32 + bf16 ----------------
__global__ void bn_relu_pack_k(const float* __restrict__ OUT, const float* __restrict__ st,
                               const float* __restrict__ g, const float* __restrict__ b,
                               float* __restrict__ X, bf16* __restrict__ Xb, int cout) {
  int idx = blockIdx.x * 256 + threadIdx.x;
  if (idx >= R * cout) return;
  int c = idx % cout;
  float v = (OUT[idx] - st[c]) * st[cout + c] * g[c] + b[c];
  v = fmaxf(v, 0.f);
  X[idx] = v;  Xb[idx] = (bf16)v;
}

// ---------------- (n,t,V,c) -> (n,c,t,V) final transpose ----------------
__global__ void transpose_out_k(const float* __restrict__ X, float* __restrict__ out) {
  size_t idx = (size_t)blockIdx.x * 256 + threadIdx.x;
  if (idx >= (size_t)NB * 128 * TT * V) return;
  int v = idx % V; size_t q = idx / V;
  int t = q % TT;  q /= TT;
  int c = q % 128; int n = (int)(q / 128);
  out[idx] = X[(((size_t)((n * TT + t) * V + v)) * 128) + c];
}

// ---------------- host-side layer ----------------
static void run_layer(hipStream_t stream, int cin, int cout,
                      float* X, bf16* Xb, bf16* XS, float* Y, float* D, float* OUT,
                      bf16* XR, bf16* XG, const float* G,
                      float* ST1, float* ST2, float* ST3,
                      const float* fm, const float* lw, const float* lb,
                      const float* bn_g, const float* bn_b,
                      const float* w_w, const float* w1_w, const float* w1_b,
                      const float* bns_g, const float* bns_b,
                      const float* dw, const float* db,
                      const float* dbn_g, const float* dbn_b) {
  shift_mask_k<<<(R * cin + 255) / 256, 256, 0, stream>>>(X, fm, XS, cin);
  gemm_bf16_k<<<dim3(R / 128, cout / 16), 256, 0, stream>>>(
      XS, nullptr, lw, nullptr, 0, 0, lb, Y, nullptr, R, cout, cin);
  const int F = V * cout;
  hipMemsetAsync(ST1, 0, 2 * F * sizeof(float), stream);
  colstats_k<<<dim3((F + 255) / 256, (NT + 511) / 512), 256, 0, stream>>>(Y, ST1, F, NT);
  stats_fin_k<<<(F + 255) / 256, 256, 0, stream>>>(ST1, F, 1.0f / NT);
  if (dw) {
    gemm_bf16_k<<<dim3(R / 128, cout / 16), 256, 0, stream>>>(
        Xb, nullptr, dw, nullptr, 1, 0, db, D, nullptr, R, cout, cin);
    hipMemsetAsync(ST3, 0, 2 * cout * sizeof(float), stream);
    colstats_k<<<dim3((cout + 255) / 256, (R + 511) / 512), 256, 0, stream>>>(D, ST3, cout, R);
    stats_fin_k<<<1, 256, 0, stream>>>(ST3, cout, 1.0f / R);
  }
  bn_shift_res_k<<<(R * cout + 255) / 256, 256, 0, stream>>>(
      Y, ST1, bn_g, bn_b, dw ? nullptr : X, dw ? D : nullptr,
      dw ? ST3 : nullptr, dbn_g, dbn_b, XR, cout);
  xg_k<<<NT, 256, 0, stream>>>(G, XR, XG, cout);
  gemm_bf16_k<<<dim3(R / 128, cout / 16), 256, 0, stream>>>(
      XG, XR, w_w, w1_w, 1, 1, w1_b, OUT, nullptr, R, cout, cout);
  hipMemsetAsync(ST2, 0, 2 * cout * sizeof(float), stream);
  colstats_k<<<dim3((cout + 255) / 256, (R + 511) / 512), 256, 0, stream>>>(OUT, ST2, cout, R);
  stats_fin_k<<<1, 256, 0, stream>>>(ST2, cout, 1.0f / R);
  bn_relu_pack_k<<<(R * cout + 255) / 256, 256, 0, stream>>>(OUT, ST2, bns_g, bns_b, X, Xb, cout);
}

} // namespace

extern "C" void kernel_launch(void* const* d_in, const int* in_sizes, int n_in,
                              void* d_out, int out_size, void* d_ws, size_t ws_size,
                              hipStream_t stream) {
  (void)in_sizes; (void)n_in; (void)out_size; (void)ws_size;
  auto in = [&](int i) { return (const float*)d_in[i]; };
  char* ws = (char*)d_ws;

  float* X   = (float*)(ws + OFF_X);
  bf16*  Xb  = (bf16*)(ws + OFF_XB);
  bf16*  XS  = (bf16*)(ws + OFF_XS);
  bf16*  XG  = XS;                      // time-disjoint alias
  float* Y   = (float*)(ws + OFF_Y);
  float* D   = (float*)(ws + OFF_D);
  float* OUT = D;                       // time-disjoint alias
  bf16*  XR  = (bf16*)(ws + OFF_XR);
  bf16*  G1b = (bf16*)(ws + OFF_XR);    // pre-layer alias
  bf16*  G2b = XS;                      // pre-layer alias
  float* G   = (float*)(ws + OFF_G);
  float* ST1 = (float*)(ws + OFF_ST1);
  float* ST2 = (float*)(ws + OFF_ST2);
  float* ST3 = (float*)(ws + OFF_ST3);

  // pack input into ntvc (f32 + bf16)
  pack_x0_k<<<(R * 64 + 255) / 256, 256, 0, stream>>>(in(0), X, Xb);

  // attention: g1/g2 projections (bf16 WMMA GEMMs, N=32,K=64), logits, softmax
  gemm_bf16_k<<<dim3(R / 128, 2), 256, 0, stream>>>(
      Xb, nullptr, in(1), nullptr, 1, 0, in(2), nullptr, G1b, R, 32, 64);
  gemm_bf16_k<<<dim3(R / 128, 2), 256, 0, stream>>>(
      Xb, nullptr, in(3), nullptr, 1, 0, in(4), nullptr, G2b, R, 32, 64);
  glogits_k<<<NT, 128, 0, stream>>>(G1b, G2b, G);
  softmax25_k<<<(NT * V + 255) / 256, 256, 0, stream>>>(G, NT * V);

  // layer 1: 64 -> 64, identity residual
  run_layer(stream, 64, 64, X, Xb, XS, Y, D, OUT, XR, XG, G, ST1, ST2, ST3,
            in(5), in(6), in(7), in(8), in(9), in(10), in(11), in(12), in(13), in(14),
            nullptr, nullptr, nullptr, nullptr);
  // layer 2: 64 -> 128, conv+BN residual
  run_layer(stream, 64, 128, X, Xb, XS, Y, D, OUT, XR, XG, G, ST1, ST2, ST3,
            in(15), in(16), in(17), in(18), in(19), in(20), in(21), in(22), in(23), in(24),
            in(25), in(26), in(27), in(28));
  // layer 3: 128 -> 128, identity residual
  run_layer(stream, 128, 128, X, Xb, XS, Y, D, OUT, XR, XG, G, ST1, ST2, ST3,
            in(29), in(30), in(31), in(32), in(33), in(34), in(35), in(36), in(37), in(38),
            nullptr, nullptr, nullptr, nullptr);

  // (n,t,V,c) -> (n,c,t,V)
  transpose_out_k<<<((size_t)NB * 128 * TT * V + 255) / 256, 256, 0, stream>>>(X, (float*)d_out);
}
